// ShiftedWindowAttention_81003083203060
// MI455X (gfx1250) — matrix operand order repaired and checked by reading, not compiled, not run on hardware
//
#include <hip/hip_runtime.h>
#include <hip/hip_bf16.h>

typedef _Float16 half_t;
typedef __attribute__((ext_vector_type(16))) _Float16 v16h;
typedef __attribute__((ext_vector_type(8)))  float    v8f;

#define B_      32
#define HW_     56
#define C_      384
#define HEADS_  12
#define HD_     32
#define WS_     7
#define SHIFT_  3
#define WIN_    49
#define L_      3136                   // 56*56
#define M_      100352                 // 32*3136
#define N_QKV   1152
#define PART_   38535168ull            // 32*12*3136*32 halfs per q/k/v part

// ---------------- WMMA fragment helpers (ISA 7.12.2 layouts) ----------------

__device__ __forceinline__ v16h frag_a(const half_t* s, int lane, int ld) {
  // 16x32 f16 A: lane l -> row l&15 ; K = {kb..kb+7, kb+16..kb+23}, kb = 8*(l>>4)
  int m  = lane & 15;
  int kb = (lane & 16) ? 8 : 0;
  const half_t* p = s + m * ld + kb;
  v16h r;
#pragma unroll
  for (int h = 0; h < 8; ++h) r[h] = p[h];
#pragma unroll
  for (int h = 0; h < 8; ++h) r[8 + h] = p[16 + h];
  return r;
}

__device__ __forceinline__ v16h frag_b(const half_t* s, int lane, int ld) {
  // 32x16 f16 B from n-major storage s[n][k]: lane l -> col l&15 ; K = {kb..kb+7, kb+16..kb+23}, kb = 8*(l>>4)
  int n  = lane & 15;
  int kb = (lane & 16) ? 8 : 0;
  const half_t* p = s + n * ld + kb;
  v16h r;
#pragma unroll
  for (int h = 0; h < 8; ++h) r[h] = p[h];
#pragma unroll
  for (int h = 0; h < 8; ++h) r[8 + h] = p[16 + h];
  return r;
}

__device__ __forceinline__ v8f wmma16(v16h a, v16h b, v8f c) {
  return __builtin_amdgcn_wmma_f32_16x16x32_f16(false, a, false, b, (short)0, c,
                                                false, false);
}

// ---------------- Kernel 1: QKV GEMM (fp32 -> f16, head-major scatter) -------

__global__ __launch_bounds__(256) void qkv_gemm_kernel(
    const float* __restrict__ x, const float* __restrict__ w,
    const float* __restrict__ bias, half_t* __restrict__ qkv) {
  __shared__ half_t As[2][128 * 40];
  __shared__ half_t Bs[2][128 * 40];

  const int tid = threadIdx.x;
  const int m0 = blockIdx.x * 128, n0 = blockIdx.y * 128;
  const int ar = tid >> 1, ac = (tid & 1) * 16;          // A tile: 128 x 32
  const int bn = tid & 127, bk = (tid >> 7) * 16;        // B tile: 32 x 128 (stored n-major)

  float areg[16], breg[16];
  const int lane = tid & 31, wv = tid >> 5;
  const int wm = (wv >> 1) * 32, wn = (wv & 1) * 64;

  v8f zero = {0.f, 0.f, 0.f, 0.f, 0.f, 0.f, 0.f, 0.f};
  v8f acc[2][4];
#pragma unroll
  for (int i = 0; i < 2; ++i)
#pragma unroll
    for (int j = 0; j < 4; ++j) acc[i][j] = zero;

  auto gload = [&](int k0) {
#pragma unroll
    for (int i = 0; i < 4; ++i) {
      float4 t = *(const float4*)(x + (size_t)(m0 + ar) * C_ + k0 + ac + i * 4);
      areg[i * 4 + 0] = t.x; areg[i * 4 + 1] = t.y;
      areg[i * 4 + 2] = t.z; areg[i * 4 + 3] = t.w;
    }
#pragma unroll
    for (int i = 0; i < 16; ++i)
      breg[i] = w[(size_t)(k0 + bk + i) * N_QKV + n0 + bn];
  };
  auto sstore = [&](int buf) {
#pragma unroll
    for (int i = 0; i < 16; ++i) As[buf][ar * 40 + ac + i] = (half_t)areg[i];
#pragma unroll
    for (int i = 0; i < 16; ++i) Bs[buf][bn * 40 + bk + i] = (half_t)breg[i];
  };

  gload(0);
  sstore(0);
  __syncthreads();

  for (int kt = 0; kt < 12; ++kt) {
    int cur = kt & 1;
    if (kt + 1 < 12) gload((kt + 1) * 32);
    v16h a0 = frag_a(&As[cur][wm * 40], lane, 40);
    v16h a1 = frag_a(&As[cur][(wm + 16) * 40], lane, 40);
#pragma unroll
    for (int j = 0; j < 4; ++j) {
      v16h bf = frag_b(&Bs[cur][(wn + j * 16) * 40], lane, 40);
      acc[0][j] = wmma16(a0, bf, acc[0][j]);
      acc[1][j] = wmma16(a1, bf, acc[1][j]);
    }
    if (kt + 1 < 12) sstore(cur ^ 1);
    __syncthreads();
  }

  const int ln = lane & 15, hi = lane >> 4;
#pragma unroll
  for (int i = 0; i < 2; ++i) {
#pragma unroll
    for (int j = 0; j < 4; ++j) {
      int gn = n0 + wn + j * 16 + ln;
      int part = gn / C_, rem = gn % C_;
      int head = rem >> 5, d = rem & 31;
      float bv = bias[gn];
#pragma unroll
      for (int jj = 0; jj < 8; ++jj) {
        int gm = m0 + wm + i * 16 + jj + hi * 8;
        int bb = gm / L_, pos = gm % L_;
        size_t dst = (size_t)part * PART_ +
                     (((size_t)(bb * HEADS_ + head)) * L_ + pos) * HD_ + d;
        qkv[dst] = (half_t)(acc[i][j][jj] + bv);
      }
    }
  }
}

// ---------------- Kernel 2: windowed attention (one block per window*head) ---

__global__ __launch_bounds__(128) void attn_kernel(
    const half_t* __restrict__ q, const half_t* __restrict__ k,
    const half_t* __restrict__ v, const float* __restrict__ btab,
    half_t* __restrict__ aout) {
  __shared__ half_t Qs[64 * 40];   // [token][d]
  __shared__ half_t Ks[64 * 40];   // [token][d]  (== n-major B for Q*K^T)
  __shared__ half_t Vt[32 * 72];   // [d][token]  (== n-major B for P*V)
  __shared__ half_t Ps[64 * 72];   // [token][token] probabilities (f16)

  const int tid = threadIdx.x;
  const int head = blockIdx.x % HEADS_;
  const int win = blockIdx.x / HEADS_;
  const int bb = win >> 6, wrem = win & 63;
  const int wh = wrem >> 3, ww = wrem & 7;

  // zero V padding region (avoid 0*NaN in P@V); Q/K pad rows are harmless
  for (int i = tid; i < (32 * 72) / 2; i += 128) ((unsigned int*)Vt)[i] = 0u;

  const size_t base = (((size_t)(bb * HEADS_ + head)) * L_) * HD_;
  if (tid < 2 * WIN_) {
    int tok = tid >> 1, off = (tid & 1) * 16;
    int i = tok / WS_, j = tok - i * WS_;
    int hs = wh * WS_ + i + SHIFT_; if (hs >= HW_) hs -= HW_;
    int wsx = ww * WS_ + j + SHIFT_; if (wsx >= HW_) wsx -= HW_;
    size_t src = base + (size_t)(hs * HW_ + wsx) * HD_ + off;
    uint4 qv = *(const uint4*)(q + src);
    uint4 kv = *(const uint4*)(k + src);
    *(uint4*)&Qs[tok * 40 + off] = qv;
    *(uint4*)&Qs[tok * 40 + off + 8] = *(((const uint4*)(q + src)) + 1);
    // (two 8-half chunks each)
    *(uint4*)&Ks[tok * 40 + off] = kv;
    *(uint4*)&Ks[tok * 40 + off + 8] = *(((const uint4*)(k + src)) + 1);
    const half_t* vr = v + src;
#pragma unroll
    for (int d = 0; d < 16; ++d) Vt[(off + d) * 72 + tok] = vr[d];
  }
  __syncthreads();

  const int lane = tid & 31, wv = tid >> 5;
  const int ln = lane & 15, hi = lane >> 4;

  // S = Q * K^T  (each wave: 16 query rows x 64 key cols)
  v8f zero = {0.f, 0.f, 0.f, 0.f, 0.f, 0.f, 0.f, 0.f};
  v16h aq = frag_a(&Qs[(wv * 16) * 40], lane, 40);
  v8f sacc[4];
#pragma unroll
  for (int j = 0; j < 4; ++j) {
    v16h bf = frag_b(&Ks[(j * 16) * 40], lane, 40);
    sacc[j] = wmma16(aq, bf, zero);
  }

  const float scale = 0.17677669529663687f;  // 1/sqrt(32)
  const float* bptr = btab + head;

  // scale + relative-position bias + masked softmax, write P (f16) to LDS
#pragma unroll
  for (int jj = 0; jj < 8; ++jj) {
    int row = wv * 16 + jj + 8 * hi;
    int i1 = row / WS_, j1 = row - i1 * WS_;
    float sv[4];
#pragma unroll
    for (int ct = 0; ct < 4; ++ct) {
      int col = ct * 16 + ln;
      float val = -1e30f;
      if (row < WIN_ && col < WIN_) {
        int i2 = col / WS_, j2 = col - i2 * WS_;
        int idx = (i1 - i2 + WS_ - 1) * (2 * WS_ - 1) + (j1 - j2 + WS_ - 1);
        val = sacc[ct][jj] * scale + bptr[idx * HEADS_];
      }
      sv[ct] = val;
    }
    float m = fmaxf(fmaxf(sv[0], sv[1]), fmaxf(sv[2], sv[3]));
#pragma unroll
    for (int msk = 1; msk < 16; msk <<= 1) m = fmaxf(m, __shfl_xor(m, msk));
    float e[4], s = 0.f;
#pragma unroll
    for (int ct = 0; ct < 4; ++ct) { e[ct] = __expf(sv[ct] - m); s += e[ct]; }
#pragma unroll
    for (int msk = 1; msk < 16; msk <<= 1) s += __shfl_xor(s, msk);
    float inv = 1.f / s;
#pragma unroll
    for (int ct = 0; ct < 4; ++ct)
      Ps[row * 72 + ct * 16 + ln] = (half_t)(e[ct] * inv);
  }
  __syncthreads();

  // O = P * V (each wave: its 16 rows x 32 dims, K = 64 in two steps)
  v16h pa0 = frag_a(&Ps[(wv * 16) * 72], lane, 72);
  v16h pa1 = frag_a(&Ps[(wv * 16) * 72 + 32], lane, 72);
  v8f oacc[2];
#pragma unroll
  for (int nj = 0; nj < 2; ++nj) {
    v16h b0 = frag_b(&Vt[(nj * 16) * 72], lane, 72);
    v16h b1 = frag_b(&Vt[(nj * 16) * 72 + 32], lane, 72);
    oacc[nj] = wmma16(pa0, b0, zero);
    oacc[nj] = wmma16(pa1, b1, oacc[nj]);
  }

  // scatter back with inverse roll into natural [b][pos][C] f16 buffer
#pragma unroll
  for (int nj = 0; nj < 2; ++nj) {
#pragma unroll
    for (int jj = 0; jj < 8; ++jj) {
      int t = wv * 16 + jj + 8 * hi;
      if (t < WIN_) {
        int i = t / WS_, j = t - i * WS_;
        int hs = wh * WS_ + i + SHIFT_; if (hs >= HW_) hs -= HW_;
        int wsx = ww * WS_ + j + SHIFT_; if (wsx >= HW_) wsx -= HW_;
        int d = nj * 16 + ln;
        size_t dst = ((size_t)bb * L_ + hs * HW_ + wsx) * C_ + head * HD_ + d;
        aout[dst] = (half_t)oacc[nj][jj];
      }
    }
  }
}

// ---------------- Kernel 3: output projection (f16 A x fp32 W -> fp32) -------

__global__ __launch_bounds__(256) void proj_gemm_kernel(
    const half_t* __restrict__ a, const float* __restrict__ w,
    const float* __restrict__ bias, float* __restrict__ out) {
  __shared__ half_t As[2][128 * 40];
  __shared__ half_t Bs[2][128 * 40];

  const int tid = threadIdx.x;
  const int m0 = blockIdx.x * 128, n0 = blockIdx.y * 128;
  const int ar = tid >> 1, ac = (tid & 1) * 16;
  const int bn = tid & 127, bk = (tid >> 7) * 16;

  uint4 areg0, areg1;
  float breg[16];
  const int lane = tid & 31, wv = tid >> 5;
  const int wm = (wv >> 1) * 32, wn = (wv & 1) * 64;

  v8f zero = {0.f, 0.f, 0.f, 0.f, 0.f, 0.f, 0.f, 0.f};
  v8f acc[2][4];
#pragma unroll
  for (int i = 0; i < 2; ++i)
#pragma unroll
    for (int j = 0; j < 4; ++j) acc[i][j] = zero;

  auto gload = [&](int k0) {
    const uint4* p = (const uint4*)(a + (size_t)(m0 + ar) * C_ + k0 + ac);
    areg0 = p[0];
    areg1 = p[1];
#pragma unroll
    for (int i = 0; i < 16; ++i)
      breg[i] = w[(size_t)(k0 + bk + i) * C_ + n0 + bn];
  };
  auto sstore = [&](int buf) {
    *(uint4*)&As[buf][ar * 40 + ac] = areg0;
    *(uint4*)&As[buf][ar * 40 + ac + 8] = areg1;
#pragma unroll
    for (int i = 0; i < 16; ++i) Bs[buf][bn * 40 + bk + i] = (half_t)breg[i];
  };

  gload(0);
  sstore(0);
  __syncthreads();

  for (int kt = 0; kt < 12; ++kt) {
    int cur = kt & 1;
    if (kt + 1 < 12) gload((kt + 1) * 32);
    v16h a0 = frag_a(&As[cur][wm * 40], lane, 40);
    v16h a1 = frag_a(&As[cur][(wm + 16) * 40], lane, 40);
#pragma unroll
    for (int j = 0; j < 4; ++j) {
      v16h bf = frag_b(&Bs[cur][(wn + j * 16) * 40], lane, 40);
      acc[0][j] = wmma16(a0, bf, acc[0][j]);
      acc[1][j] = wmma16(a1, bf, acc[1][j]);
    }
    if (kt + 1 < 12) sstore(cur ^ 1);
    __syncthreads();
  }

  const int ln = lane & 15, hi = lane >> 4;
#pragma unroll
  for (int i = 0; i < 2; ++i) {
#pragma unroll
    for (int j = 0; j < 4; ++j) {
      int gn = n0 + wn + j * 16 + ln;
      float bv = bias[gn];
#pragma unroll
      for (int jj = 0; jj < 8; ++jj) {
        int gm = m0 + wm + i * 16 + jj + hi * 8;
        out[(size_t)gm * C_ + gn] = acc[i][j][jj] + bv;
      }
    }
  }
}

// ---------------- Host launch ------------------------------------------------

extern "C" void kernel_launch(void* const* d_in, const int* in_sizes, int n_in,
                              void* d_out, int out_size, void* d_ws, size_t ws_size,
                              hipStream_t stream) {
  const float* x      = (const float*)d_in[0];
  const float* qkv_w  = (const float*)d_in[1];
  const float* qkv_b  = (const float*)d_in[2];
  const float* proj_w = (const float*)d_in[3];
  const float* proj_b = (const float*)d_in[4];
  const float* rel    = (const float*)d_in[5];
  (void)in_sizes; (void)n_in; (void)out_size; (void)ws_size;

  half_t* qkv = (half_t*)d_ws;                 // 3 * PART_ halfs (q, k, v)
  half_t* qh  = qkv;
  half_t* kh  = qkv + PART_;
  half_t* vh  = qkv + 2 * PART_;
  half_t* ah  = qkv + 3 * PART_;               // attention output, natural order

  qkv_gemm_kernel<<<dim3(M_ / 128, N_QKV / 128), 256, 0, stream>>>(
      x, qkv_w, qkv_b, qkv);
  attn_kernel<<<dim3(2048 * HEADS_), 128, 0, stream>>>(qh, kh, vh, rel, ah);
  proj_gemm_kernel<<<dim3(M_ / 128, C_ / 128), 256, 0, stream>>>(
      ah, proj_w, proj_b, (float*)d_out);
}
